// Model_53996328845374
// MI455X (gfx1250) — matrix operand order repaired and checked by reading, not compiled, not run on hardware
//
#include <hip/hip_runtime.h>
#include <hip/hip_bf16.h>

// ---------------- problem constants (match reference) ----------------
#define NPD   100000      // num pdrug nodes
#define NSE   10000       // num seffect nodes
#define FPDIM 512
#define FSDIM 768
#define HID   64
#define NEDGE 1600000
#define NLAB  1000000

typedef __attribute__((ext_vector_type(16))) __bf16 v16bf;
typedef __attribute__((ext_vector_type(8)))  float  v8f;

// ---------------- WMMA helpers (bf16 -> f32 accumulate) ----------------
__device__ inline v8f wmma_bf16(v16bf a, v16bf b, v8f c) {
  // 8 args: (neg_a, A, neg_b, B, c_mod, C, reuse_a, reuse_b)
  return __builtin_amdgcn_wmma_f32_16x16x32_bf16(false, a, false, b, (short)0, c,
                                                 false, false);
}

// Load 16x32 A fragment (row-major f32 source, optional per-row scale) per ISA
// layout: lanes 0-15 row=lane (+0 K-base), lanes 16-31 row=lane-16 (+8 K-base);
// elements 0-7 -> K=base+e, elements 8-15 -> K=base+16+(e-8).
// Per-lane K runs are contiguous -> 4x float4 (global_load_b128).
template <bool SCALED>
__device__ inline v16bf load_A(const float* __restrict__ X, int row0, int k0,
                               int ld, const float* __restrict__ scale, int lane) {
  const int r  = row0 + (lane & 15);
  const int kb = k0 + ((lane & 16) >> 1);        // +8 for upper half-wave
  float s = 1.0f;
  if constexpr (SCALED) s = scale[r];
  const float4* p = (const float4*)(X + (size_t)r * ld + kb);  // 16B aligned
  const float4 q0 = p[0];   // K = base + 0..3
  const float4 q1 = p[1];   // K = base + 4..7
  const float4 q2 = p[4];   // K = base + 16..19
  const float4 q3 = p[5];   // K = base + 20..23
  v16bf a;
  a[0]  = (__bf16)(q0.x * s); a[1]  = (__bf16)(q0.y * s);
  a[2]  = (__bf16)(q0.z * s); a[3]  = (__bf16)(q0.w * s);
  a[4]  = (__bf16)(q1.x * s); a[5]  = (__bf16)(q1.y * s);
  a[6]  = (__bf16)(q1.z * s); a[7]  = (__bf16)(q1.w * s);
  a[8]  = (__bf16)(q2.x * s); a[9]  = (__bf16)(q2.y * s);
  a[10] = (__bf16)(q2.z * s); a[11] = (__bf16)(q2.w * s);
  a[12] = (__bf16)(q3.x * s); a[13] = (__bf16)(q3.y * s);
  a[14] = (__bf16)(q3.z * s); a[15] = (__bf16)(q3.w * s);
  return a;
}

// B fragments are pre-packed fragment-major in bf16:
//   pk[(((kblk*4 + nblk)*32 + lane)*16 + e] = bf16(W[kblk*32+((lane&16)>>1)+(e&7)+((e&8)<<1)][nblk*16+(lane&15)])
// so each lane loads its 16 elements as one contiguous 32-byte vector.
__device__ inline v16bf load_B_pk(const __bf16* __restrict__ Wpk, int kblk,
                                  int nblk, int lane) {
  const size_t idx = ((size_t)((kblk * 4 + nblk) * 32 + lane)) << 4;
  return *(const v16bf*)(Wpk + idx);
}

// Store 16x16 C/D tile: VGPR v -> row = row0 + v + (lane>=16 ? 8 : 0),
// col = n0 + (lane&15). Compile-time bias-extra/relu epilogue (no branches).
template <bool HAS_EXTRA, bool RELU>
__device__ inline void store_tile(float* __restrict__ out, v8f acc, int row0,
                                  int n0, int lane,
                                  const float* __restrict__ bias,
                                  const float* __restrict__ extra) {
  const int col   = n0 + (lane & 15);
  const int rbase = row0 + ((lane & 16) >> 1);
  const float bv = bias[col];
#pragma unroll
  for (int v = 0; v < 8; ++v) {
    const int r = rbase + v;
    float val = acc[v] + bv;
    if constexpr (HAS_EXTRA) val += extra[(size_t)r * HID + col];
    if constexpr (RELU)      val = fmaxf(val, 0.0f);
    out[(size_t)r * HID + col] = val;
  }
}

// ---------------- kernels ----------------

__global__ void fill_zero(float* __restrict__ p, int n) {
  int i = blockIdx.x * blockDim.x + threadIdx.x;
  if (i < n) p[i] = 0.0f;
}

// Pack row-major f32 W[K][64] into WMMA B-fragment-major bf16 (layout above).
__global__ void pack_B(const float* __restrict__ W, __bf16* __restrict__ out,
                       int K) {
  int t = blockIdx.x * blockDim.x + threadIdx.x;
  if (t >= K * HID) return;
  const int e    = t & 15;
  const int lane = (t >> 4) & 31;
  const int blk  = t >> 9;          // kblk*4 + nblk
  const int nblk = blk & 3;
  const int kblk = blk >> 2;
  const int k = (kblk << 5) + ((lane & 16) >> 1) + (e & 7) + ((e & 8) << 1);
  const int n = (nblk << 4) + (lane & 15);
  out[t] = (__bf16)W[(size_t)k * HID + n];
}

__global__ void count_deg(const int* __restrict__ idx, float* __restrict__ deg,
                          int n) {
  int e = blockIdx.x * blockDim.x + threadIdx.x;
  if (e < n) unsafeAtomicAdd(&deg[idx[e]], 1.0f);
}

__global__ void inv_clip(const float* __restrict__ deg, float* __restrict__ inv,
                         int n) {
  int i = blockIdx.x * blockDim.x + threadIdx.x;
  if (i < n) inv[i] = 1.0f / fmaxf(deg[i], 1.0f);
}

// h = X @ W + bias + emb     (M x K) @ (K x 64); one wave per 16x64 tile.
// X is the one big HBM stream -> prefetch one 32-wide k-tile ahead.
__global__ __launch_bounds__(256)
void gemm_bias_emb(const float* __restrict__ X, const __bf16* __restrict__ Wpk,
                   const float* __restrict__ bias, const float* __restrict__ emb,
                   float* __restrict__ out, int M, int K) {
  const int lane = threadIdx.x & 31;
  const int tile = blockIdx.x * 8 + (threadIdx.x >> 5);  // wave-uniform
  if (tile >= (M >> 4)) return;                          // whole wave exits
  const int row0 = tile << 4;
  const float* xrow = X + (size_t)(row0 + (lane & 15)) * K + ((lane & 16) >> 1);
  v8f acc0{}, acc1{}, acc2{}, acc3{};
  for (int k0 = 0; k0 < K; k0 += 32) {
    if (k0 + 32 < K)   // global_prefetch_b8: pull next k-tile of this row
      __builtin_prefetch((const void*)(xrow + k0 + 32), 0, 3);
    const int kblk = k0 >> 5;
    v16bf a = load_A<false>(X, row0, k0, K, nullptr, lane);
    acc0 = wmma_bf16(a, load_B_pk(Wpk, kblk, 0, lane), acc0);
    acc1 = wmma_bf16(a, load_B_pk(Wpk, kblk, 1, lane), acc1);
    acc2 = wmma_bf16(a, load_B_pk(Wpk, kblk, 2, lane), acc2);
    acc3 = wmma_bf16(a, load_B_pk(Wpk, kblk, 3, lane), acc3);
  }
  store_tile<true, false>(out, acc0, row0,  0, lane, bias, emb);
  store_tile<true, false>(out, acc1, row0, 16, lane, bias, emb);
  store_tile<true, false>(out, acc2, row0, 32, lane, bias, emb);
  store_tile<true, false>(out, acc3, row0, 48, lane, bias, emb);
}

// agg[dst] += xsrc[src]   (4 threads per edge, 16 channels each; float4 reads,
// per-channel global_atomic_add_f32; agg pages live in 192MB L2)
__global__ void scatter_add(const float* __restrict__ xsrc,
                            const int* __restrict__ sidx,
                            const int* __restrict__ didx,
                            float* __restrict__ agg) {
  long t = (long)blockIdx.x * blockDim.x + threadIdx.x;
  int e = (int)(t >> 2);
  if (e >= NEDGE) return;
  const int part = ((int)t & 3) << 4;
  const int s = sidx[e], d = didx[e];
  const float4* xp = (const float4*)(xsrc + (size_t)s * HID + part);
  float* ap = agg + (size_t)d * HID + part;
#pragma unroll
  for (int q = 0; q < 4; ++q) {
    float4 v = xp[q];
    unsafeAtomicAdd(ap + q * 4 + 0, v.x);
    unsafeAtomicAdd(ap + q * 4 + 1, v.y);
    unsafeAtomicAdd(ap + q * 4 + 2, v.z);
    unsafeAtomicAdd(ap + q * 4 + 3, v.w);
  }
}

// out = relu?( (agg * inv_deg) @ Wl + xdst @ Wr + bl ), RELU compile-time.
template <bool RELU>
__global__ __launch_bounds__(256)
void sage_combine(const float* __restrict__ agg, const float* __restrict__ invd,
                  const float* __restrict__ xdst,
                  const __bf16* __restrict__ WlPk, const float* __restrict__ bl,
                  const __bf16* __restrict__ WrPk,
                  float* __restrict__ out, int M) {
  const int lane = threadIdx.x & 31;
  const int tile = blockIdx.x * 8 + (threadIdx.x >> 5);
  if (tile >= (M >> 4)) return;
  const int row0 = tile << 4;
  v8f acc0{}, acc1{}, acc2{}, acc3{};
#pragma unroll
  for (int k0 = 0; k0 < HID; k0 += 32) {               // mean @ Wl
    const int kblk = k0 >> 5;
    v16bf a = load_A<true>(agg, row0, k0, HID, invd, lane);
    acc0 = wmma_bf16(a, load_B_pk(WlPk, kblk, 0, lane), acc0);
    acc1 = wmma_bf16(a, load_B_pk(WlPk, kblk, 1, lane), acc1);
    acc2 = wmma_bf16(a, load_B_pk(WlPk, kblk, 2, lane), acc2);
    acc3 = wmma_bf16(a, load_B_pk(WlPk, kblk, 3, lane), acc3);
  }
#pragma unroll
  for (int k0 = 0; k0 < HID; k0 += 32) {               // x_dst @ Wr
    const int kblk = k0 >> 5;
    v16bf a = load_A<false>(xdst, row0, k0, HID, nullptr, lane);
    acc0 = wmma_bf16(a, load_B_pk(WrPk, kblk, 0, lane), acc0);
    acc1 = wmma_bf16(a, load_B_pk(WrPk, kblk, 1, lane), acc1);
    acc2 = wmma_bf16(a, load_B_pk(WrPk, kblk, 2, lane), acc2);
    acc3 = wmma_bf16(a, load_B_pk(WrPk, kblk, 3, lane), acc3);
  }
  store_tile<false, RELU>(out, acc0, row0,  0, lane, bl, nullptr);
  store_tile<false, RELU>(out, acc1, row0, 16, lane, bl, nullptr);
  store_tile<false, RELU>(out, acc2, row0, 32, lane, bl, nullptr);
  store_tile<false, RELU>(out, acc3, row0, 48, lane, bl, nullptr);
}

// pred[e] = dot(pd2[i0], se2[i1]); also writes edge_label into out[NLAB..)
__global__ void edge_pred(const float* __restrict__ pd2,
                          const float* __restrict__ se2,
                          const int* __restrict__ eli,
                          const float* __restrict__ label,
                          float* __restrict__ out) {
  int e = blockIdx.x * blockDim.x + threadIdx.x;
  if (e >= NLAB) return;
  const int i0 = eli[e];
  const int i1 = eli[NLAB + e];
  const float4* a = (const float4*)(pd2 + (size_t)i0 * HID);
  const float4* b = (const float4*)(se2 + (size_t)i1 * HID);
  float s = 0.0f;
#pragma unroll
  for (int i = 0; i < 16; ++i) {
    float4 x = a[i], y = b[i];
    s += x.x * y.x + x.y * y.y + x.z * y.z + x.w * y.w;
  }
  out[e] = s;
  out[NLAB + e] = label[e];
}

// ---------------- host launcher ----------------
extern "C" void kernel_launch(void* const* d_in, const int* in_sizes, int n_in,
                              void* d_out, int out_size, void* d_ws, size_t ws_size,
                              hipStream_t stream) {
  (void)in_sizes; (void)n_in; (void)out_size; (void)ws_size;

  const float* x_pd   = (const float*)d_in[0];
  const float* x_se   = (const float*)d_in[1];
  // d_in[2], d_in[3]: node_id arange -> emb[node_id] == emb, unused
  const int*   eidx   = (const int*)d_in[4];   // [2, NEDGE]: src row0, dst row1
  const int*   eli    = (const int*)d_in[5];   // [2, NLAB]
  const float* label  = (const float*)d_in[6];
  const float* W_pd   = (const float*)d_in[7];
  const float* b_pd   = (const float*)d_in[8];
  const float* W_se   = (const float*)d_in[9];
  const float* b_se   = (const float*)d_in[10];
  const float* emb_pd = (const float*)d_in[11];
  const float* emb_se = (const float*)d_in[12];
  const float* c1p_Wl = (const float*)d_in[13];
  const float* c1p_bl = (const float*)d_in[14];
  const float* c1p_Wr = (const float*)d_in[15];
  const float* c1s_Wl = (const float*)d_in[16];
  const float* c1s_bl = (const float*)d_in[17];
  const float* c1s_Wr = (const float*)d_in[18];
  const float* c2p_Wl = (const float*)d_in[19];
  const float* c2p_bl = (const float*)d_in[20];
  const float* c2p_Wr = (const float*)d_in[21];
  const float* c2s_Wl = (const float*)d_in[22];
  const float* c2s_bl = (const float*)d_in[23];
  const float* c2s_Wr = (const float*)d_in[24];

  const int* src = eidx;             // pdrug index per edge
  const int* dst = eidx + NEDGE;     // seffect index per edge

  // workspace carve (floats); node buffers ~27.8M floats, + ~230KB packed W
  float* ws     = (float*)d_ws;
  float* h_pd   = ws;                       // NPD*HID
  float* h_se   = h_pd + (size_t)NPD * HID; // NSE*HID
  float* pd1    = h_se + (size_t)NSE * HID;
  float* se1    = pd1  + (size_t)NPD * HID;
  float* pd2    = se1  + (size_t)NSE * HID;
  float* se2    = pd2  + (size_t)NPD * HID;
  float* agg    = se2  + (size_t)NSE * HID; // NPD*HID (max), reused 4x
  float* deg_pd = agg  + (size_t)NPD * HID; // NPD
  float* deg_se = deg_pd + NPD;             // NSE
  float* inv_pd = deg_se + NSE;             // NPD
  float* inv_se = inv_pd + NPD;             // NSE

  // packed bf16 weights (fragment-major); offsets all >=32B aligned
  __bf16* pk_Wpd = (__bf16*)(inv_se + NSE);        // FPDIM*HID
  __bf16* pk_Wse = pk_Wpd + (size_t)FPDIM * HID;   // FSDIM*HID
  __bf16* pk_c1p_Wl = pk_Wse + (size_t)FSDIM * HID;
  __bf16* pk_c1p_Wr = pk_c1p_Wl + HID * HID;
  __bf16* pk_c1s_Wl = pk_c1p_Wr + HID * HID;
  __bf16* pk_c1s_Wr = pk_c1s_Wl + HID * HID;
  __bf16* pk_c2p_Wl = pk_c1s_Wr + HID * HID;
  __bf16* pk_c2p_Wr = pk_c2p_Wl + HID * HID;
  __bf16* pk_c2s_Wl = pk_c2p_Wr + HID * HID;
  __bf16* pk_c2s_Wr = pk_c2s_Wl + HID * HID;

  float* out = (float*)d_out;

  const int B = 256;
  const int g_np   = (NPD + B - 1) / B;
  const int g_ns   = (NSE + B - 1) / B;
  const int g_edge = (NEDGE + B - 1) / B;
  const int g_scat = (NEDGE * 4 + B - 1) / B;
  const int g_aggp = (NPD * HID + B - 1) / B;
  const int g_aggs = (NSE * HID + B - 1) / B;
  const int g_lab  = (NLAB + B - 1) / B;
  const int g_gpd  = ((NPD / 16) + 7) / 8;   // waves of 16-row tiles, 8/block
  const int g_gse  = ((NSE / 16) + 7) / 8;
  const int g_pkin = (FPDIM * HID + B - 1) / B;
  const int g_pkse = (FSDIM * HID + B - 1) / B;
  const int g_pkh  = (HID * HID + B - 1) / B;

  // pack all weight matrices to bf16 fragment layout (once per launch)
  pack_B<<<g_pkin, B, 0, stream>>>(W_pd, pk_Wpd, FPDIM);
  pack_B<<<g_pkse, B, 0, stream>>>(W_se, pk_Wse, FSDIM);
  pack_B<<<g_pkh, B, 0, stream>>>(c1p_Wl, pk_c1p_Wl, HID);
  pack_B<<<g_pkh, B, 0, stream>>>(c1p_Wr, pk_c1p_Wr, HID);
  pack_B<<<g_pkh, B, 0, stream>>>(c1s_Wl, pk_c1s_Wl, HID);
  pack_B<<<g_pkh, B, 0, stream>>>(c1s_Wr, pk_c1s_Wr, HID);
  pack_B<<<g_pkh, B, 0, stream>>>(c2p_Wl, pk_c2p_Wl, HID);
  pack_B<<<g_pkh, B, 0, stream>>>(c2p_Wr, pk_c2p_Wr, HID);
  pack_B<<<g_pkh, B, 0, stream>>>(c2s_Wl, pk_c2s_Wl, HID);
  pack_B<<<g_pkh, B, 0, stream>>>(c2s_Wr, pk_c2s_Wr, HID);

  // degrees (shared by both layers per direction)
  fill_zero<<<g_np, B, 0, stream>>>(deg_pd, NPD);
  fill_zero<<<g_ns, B, 0, stream>>>(deg_se, NSE);
  count_deg<<<g_edge, B, 0, stream>>>(dst, deg_se, NEDGE);
  count_deg<<<g_edge, B, 0, stream>>>(src, deg_pd, NEDGE);
  inv_clip<<<g_np, B, 0, stream>>>(deg_pd, inv_pd, NPD);
  inv_clip<<<g_ns, B, 0, stream>>>(deg_se, inv_se, NSE);

  // fused input projection + embedding
  gemm_bias_emb<<<g_gpd, B, 0, stream>>>(x_pd, pk_Wpd, b_pd, emb_pd, h_pd, NPD, FPDIM);
  gemm_bias_emb<<<g_gse, B, 0, stream>>>(x_se, pk_Wse, b_se, emb_se, h_se, NSE, FSDIM);

  // layer 1, p2s -> se1 (relu)
  fill_zero<<<g_aggs, B, 0, stream>>>(agg, NSE * HID);
  scatter_add<<<g_scat, B, 0, stream>>>(h_pd, src, dst, agg);
  sage_combine<true><<<g_gse, B, 0, stream>>>(agg, inv_se, h_se, pk_c1p_Wl,
                                              c1p_bl, pk_c1p_Wr, se1, NSE);
  // layer 1, s2p -> pd1 (relu)
  fill_zero<<<g_aggp, B, 0, stream>>>(agg, NPD * HID);
  scatter_add<<<g_scat, B, 0, stream>>>(h_se, dst, src, agg);
  sage_combine<true><<<g_gpd, B, 0, stream>>>(agg, inv_pd, h_pd, pk_c1s_Wl,
                                              c1s_bl, pk_c1s_Wr, pd1, NPD);
  // layer 2, p2s -> se2
  fill_zero<<<g_aggs, B, 0, stream>>>(agg, NSE * HID);
  scatter_add<<<g_scat, B, 0, stream>>>(pd1, src, dst, agg);
  sage_combine<false><<<g_gse, B, 0, stream>>>(agg, inv_se, se1, pk_c2p_Wl,
                                               c2p_bl, pk_c2p_Wr, se2, NSE);
  // layer 2, s2p -> pd2
  fill_zero<<<g_aggp, B, 0, stream>>>(agg, NPD * HID);
  scatter_add<<<g_scat, B, 0, stream>>>(se1, dst, src, agg);
  sage_combine<false><<<g_gpd, B, 0, stream>>>(agg, inv_pd, pd1, pk_c2s_Wl,
                                               c2s_bl, pk_c2s_Wr, pd2, NPD);

  // classifier + label copy (tuple output: [pred, edge_label])
  edge_pred<<<g_lab, B, 0, stream>>>(pd2, se2, eli, label, out);
}
